// ArrowAttention_25683904430202
// MI455X (gfx1250) — matrix operand order repaired and checked by reading, not compiled, not run on hardware
//
#include <hip/hip_runtime.h>
#include <hip/hip_bf16.h>
#include <hip/hip_fp16.h>

typedef _Float16 v16h __attribute__((ext_vector_type(16)));
typedef _Float16 v8h  __attribute__((ext_vector_type(8)));
typedef float    v8f  __attribute__((ext_vector_type(8)));

#define D_MODEL 1024
#define N_HEADS 16
#define HEAD_DIM 64
#define SEQ_L   2048
#define WIN     128

// ---- WMMA fragment loaders (contiguous k order per lane half, wave32) ----
// A fragment 16x32 f16: lane lr holds row M=lr; g=lane>>4 selects K halves:
//   elems e -> K = kb + g*16 + e
__device__ __forceinline__ v16h load_a_frag(const _Float16* row, int kb, int g) {
  v8h lo = *(const v8h*)(row + kb + g * 16);
  v8h hi = *(const v8h*)(row + kb + g * 16 + 8);
  v16h r;
#pragma unroll
  for (int i = 0; i < 8; ++i) { r[i] = lo[i]; r[8 + i] = hi[i]; }
  return r;
}
// B fragment 32x16 f16: lane lr holds column N=lr; elems e -> K = kb + g*16 + e
// (requires memory contiguous along contraction dim: weights/V stored transposed)
__device__ __forceinline__ v16h load_b_frag(const _Float16* row, int kb, int g) {
  return *(const v16h*)(row + kb + g * 16);
}
__device__ __forceinline__ v8f wmma_f16(v16h a, v16h b, v8f c) {
  return __builtin_amdgcn_wmma_f32_16x16x32_f16(false, a, false, b, (short)0, c,
                                                false, false);
}

// ---- conversion kernels ----
__global__ void cvt_f32_to_f16(const float* __restrict__ in,
                               _Float16* __restrict__ out, int n) {
  int i = blockIdx.x * blockDim.x + threadIdx.x;
  if (i < n) out[i] = (_Float16)in[i];
}
__global__ void cvt_transpose(const float* __restrict__ in,
                              _Float16* __restrict__ out, int rows, int cols) {
  int i = blockIdx.x * blockDim.x + threadIdx.x;
  if (i < rows * cols) {
    int r = i / cols, c = i % cols;
    out[(size_t)c * rows + r] = (_Float16)in[i];
  }
}

// ---- f16 WMMA GEMM, 32x64 register-blocked per wave ----
//   C[M,N] = A[M,K] * Bt[N,K]^T + bias[n]
// mode 0: f16 out row-major; mode 1: f16 out transposed [N,M]; mode 2: f32 out
__global__ void gemm16(const _Float16* __restrict__ A,
                       const _Float16* __restrict__ Bt,
                       const float* __restrict__ bias,
                       _Float16* __restrict__ outH, float* __restrict__ outF,
                       int M, int N, int K, int mode) {
  int wave = threadIdx.x >> 5;
  int lane = threadIdx.x & 31;
  int lr = lane & 15, g = lane >> 4;
  int tilesN = N >> 6;                       // 64-wide N tiles
  int tid = blockIdx.x * (blockDim.x >> 5) + wave;
  if (tid >= (M >> 5) * tilesN) return;
  int tm = (tid / tilesN) << 5;              // 32-tall M tile
  int tn = (tid % tilesN) << 6;

  const _Float16* arow0 = A + (size_t)(tm + lr) * K;
  const _Float16* arow1 = A + (size_t)(tm + 16 + lr) * K;
  const _Float16* brow0 = Bt + (size_t)(tn + lr) * K;
  const _Float16* brow1 = Bt + (size_t)(tn + 16 + lr) * K;
  const _Float16* brow2 = Bt + (size_t)(tn + 32 + lr) * K;
  const _Float16* brow3 = Bt + (size_t)(tn + 48 + lr) * K;

  v8f acc[2][4] = {{{}, {}, {}, {}}, {{}, {}, {}, {}}};
  for (int kb = 0; kb < K; kb += 32) {
    v16h a0 = load_a_frag(arow0, kb, g);
    v16h a1 = load_a_frag(arow1, kb, g);
    v16h b0 = load_b_frag(brow0, kb, g);
    v16h b1 = load_b_frag(brow1, kb, g);
    v16h b2 = load_b_frag(brow2, kb, g);
    v16h b3 = load_b_frag(brow3, kb, g);
    acc[0][0] = wmma_f16(a0, b0, acc[0][0]);
    acc[1][0] = wmma_f16(a1, b0, acc[1][0]);
    acc[0][1] = wmma_f16(a0, b1, acc[0][1]);
    acc[1][1] = wmma_f16(a1, b1, acc[1][1]);
    acc[0][2] = wmma_f16(a0, b2, acc[0][2]);
    acc[1][2] = wmma_f16(a1, b2, acc[1][2]);
    acc[0][3] = wmma_f16(a0, b3, acc[0][3]);
    acc[1][3] = wmma_f16(a1, b3, acc[1][3]);
  }
  // C layout per 16x16 tile: lane holds column n; VGPR v -> row m = 8*g + v
#pragma unroll
  for (int mi = 0; mi < 2; ++mi) {
#pragma unroll
    for (int nj = 0; nj < 4; ++nj) {
#pragma unroll
      for (int v = 0; v < 8; ++v) {
        int m = tm + mi * 16 + 8 * g + v;
        int n = tn + nj * 16 + lr;
        float val = acc[mi][nj][v] + bias[n];
        if (mode == 0)      outH[(size_t)m * N + n] = (_Float16)val;
        else if (mode == 1) outH[(size_t)n * M + m] = (_Float16)val;
        else                outF[(size_t)m * N + n] = val;
      }
    }
  }
}

// ---- flash arrow-attention: 1 wave per (16-query block, head) ----
__global__ void arrow_attn(const _Float16* __restrict__ Qh,
                           const _Float16* __restrict__ Kh,
                           const _Float16* __restrict__ Vt,
                           const unsigned char* __restrict__ amask,
                           _Float16* __restrict__ AttnH) {
  __shared__ _Float16 pbuf[16 * 32];
  int qb = blockIdx.x;   // 0..127
  int h = blockIdx.y;    // 0..15
  int lane = threadIdx.x;
  int lr = lane & 15, g = lane >> 4;
  int hoff = h * HEAD_DIM;

  const _Float16* qrow = Qh + (size_t)(qb * 16 + lr) * D_MODEL + hoff;
  v16h qa0 = load_a_frag(qrow, 0, g);
  v16h qa1 = load_a_frag(qrow, 32, g);

  v8f accO[4] = {{}, {}, {}, {}};
  float rowmax[8], rowsum[8];
#pragma unroll
  for (int v = 0; v < 8; ++v) { rowmax[v] = -1e30f; rowsum[v] = 0.f; }

  int q0 = qb * 16;
  int jt_lo = max(0, q0 - WIN) >> 5;
  int jt_hi = min(SEQ_L - 1, q0 + 15 + WIN) >> 5;
  int extra = (jt_lo > 0) ? 1 : 0;  // prepend sink tile 0 if not in window
  int ntiles = jt_hi - jt_lo + 1 + extra;

  for (int it = 0; it < ntiles; ++it) {
    int jt = extra ? (it == 0 ? 0 : jt_lo + it - 1) : (jt_lo + it);
    int jb = jt << 5;

    const _Float16* krow0 = Kh + (size_t)(jb + lr) * D_MODEL + hoff;
    const _Float16* krow1 = Kh + (size_t)(jb + 16 + lr) * D_MODEL + hoff;
    v8f s0 = {}, s1 = {};
    s0 = wmma_f16(qa0, load_b_frag(krow0, 0, g), s0);
    s0 = wmma_f16(qa1, load_b_frag(krow0, 32, g), s0);
    s1 = wmma_f16(qa0, load_b_frag(krow1, 0, g), s1);
    s1 = wmma_f16(qa1, load_b_frag(krow1, 32, g), s1);

    int j0 = jb + lr, j1 = jb + 16 + lr;
    bool kv0 = amask[j0] != 0, kv1 = amask[j1] != 0;
    float ascale[8];
#pragma unroll
    for (int v = 0; v < 8; ++v) {
      int i = q0 + 8 * g + v;
      bool a0 = kv0 && ((j0 == 0) || (i - j0 <= WIN && j0 - i <= WIN));
      bool a1 = kv1 && ((j1 == 0) || (i - j1 <= WIN && j1 - i <= WIN));
      float x0 = a0 ? s0[v] * 0.125f : -1e30f;
      float x1 = a1 ? s1[v] * 0.125f : -1e30f;
      float t = fmaxf(x0, x1);
      t = fmaxf(t, __shfl_xor(t, 1));
      t = fmaxf(t, __shfl_xor(t, 2));
      t = fmaxf(t, __shfl_xor(t, 4));
      t = fmaxf(t, __shfl_xor(t, 8));   // reduce within 16-lane column group
      float mo = rowmax[v];
      float mn = fmaxf(mo, t);
      float alpha = __expf(mo - mn);
      float p0 = __expf(x0 - mn);
      float p1 = __expf(x1 - mn);
      float ps = p0 + p1;
      ps += __shfl_xor(ps, 1);
      ps += __shfl_xor(ps, 2);
      ps += __shfl_xor(ps, 4);
      ps += __shfl_xor(ps, 8);
      rowsum[v] = rowsum[v] * alpha + ps;
      rowmax[v] = mn;
      ascale[v] = alpha;
      int m = 8 * g + v;
      pbuf[m * 32 + lr] = (_Float16)p0;        // transpose C->A via LDS
      pbuf[m * 32 + 16 + lr] = (_Float16)p1;
    }
    __syncthreads();
    v16h pa = load_a_frag(pbuf + lr * 32, 0, g);
    __syncthreads();
#pragma unroll
    for (int nt = 0; nt < 4; ++nt) {
#pragma unroll
      for (int v = 0; v < 8; ++v) accO[nt][v] *= ascale[v];
      const _Float16* vrow = Vt + (size_t)(hoff + nt * 16 + lr) * SEQ_L;
      v16h vb = load_b_frag(vrow, jb, g);
      accO[nt] = wmma_f16(pa, vb, accO[nt]);
    }
  }

#pragma unroll
  for (int nt = 0; nt < 4; ++nt) {
#pragma unroll
    for (int v = 0; v < 8; ++v) {
      int row = q0 + 8 * g + v;
      int col = hoff + nt * 16 + lr;
      AttnH[(size_t)row * D_MODEL + col] = (_Float16)(accO[nt][v] / rowsum[v]);
    }
  }
}

// ---- row 0: full attention + logn scale (overwrites attn row 0) ----
__global__ void row0_fix(const _Float16* __restrict__ Qh,
                         const _Float16* __restrict__ Kh,
                         const _Float16* __restrict__ Vt,
                         const unsigned char* __restrict__ amask,
                         _Float16* __restrict__ AttnH) {
  __shared__ float qs[HEAD_DIM];
  __shared__ float probs[SEQ_L];
  __shared__ float red[256];
  int h = blockIdx.x;
  int t = threadIdx.x;
  int hoff = h * HEAD_DIM;
  if (t < HEAD_DIM) qs[t] = (float)Qh[hoff + t];
  __syncthreads();
  const float lognscale = 0.125f * (11.0f / 9.0f);  // (1/8) * log_512(2048)
  float lmax = -1e30f;
  for (int j = t; j < SEQ_L; j += 256) {
    const _Float16* kr = Kh + (size_t)j * D_MODEL + hoff;
    float s = 0.f;
#pragma unroll
    for (int d = 0; d < HEAD_DIM; ++d) s += qs[d] * (float)kr[d];
    s *= lognscale;
    if (amask[j] == 0) s = -1e30f;
    probs[j] = s;
    lmax = fmaxf(lmax, s);
  }
  red[t] = lmax; __syncthreads();
  for (int off = 128; off > 0; off >>= 1) {
    if (t < off) red[t] = fmaxf(red[t], red[t + off]);
    __syncthreads();
  }
  float gmax = red[0]; __syncthreads();
  float lsum = 0.f;
  for (int j = t; j < SEQ_L; j += 256) {
    float p = __expf(probs[j] - gmax);
    probs[j] = p; lsum += p;
  }
  red[t] = lsum; __syncthreads();
  for (int off = 128; off > 0; off >>= 1) {
    if (t < off) red[t] += red[t + off];
    __syncthreads();
  }
  float inv = 1.f / red[0];
  __syncthreads();
  if (t < HEAD_DIM) {
    const _Float16* vr = Vt + (size_t)(hoff + t) * SEQ_L;
    float o = 0.f;
    for (int j = 0; j < SEQ_L; ++j) o += probs[j] * (float)vr[j];
    AttnH[hoff + t] = (_Float16)(o * inv);
  }
}

extern "C" void kernel_launch(void* const* d_in, const int* in_sizes, int n_in,
                              void* d_out, int out_size, void* d_ws, size_t ws_size,
                              hipStream_t stream) {
  const float* X = (const float*)d_in[0];
  const unsigned char* amask = (const unsigned char*)d_in[1];
  const float* Wq = (const float*)d_in[2];
  const float* bq = (const float*)d_in[3];
  const float* Wk = (const float*)d_in[4];
  const float* bk = (const float*)d_in[5];
  const float* Wv = (const float*)d_in[6];
  const float* bv = (const float*)d_in[7];
  const float* Wo = (const float*)d_in[8];
  const float* bo = (const float*)d_in[9];
  float* out = (float*)d_out;

  char* ws = (char*)d_ws;
  size_t off = 0;
  auto alloc = [&](size_t bytes) { void* p = ws + off; off += bytes; return p; };
  _Float16* Xh    = (_Float16*)alloc((size_t)SEQ_L * D_MODEL * 2);
  _Float16* WqT   = (_Float16*)alloc((size_t)D_MODEL * D_MODEL * 2);
  _Float16* WkT   = (_Float16*)alloc((size_t)D_MODEL * D_MODEL * 2);
  _Float16* WvT   = (_Float16*)alloc((size_t)D_MODEL * D_MODEL * 2);
  _Float16* WoT   = (_Float16*)alloc((size_t)D_MODEL * D_MODEL * 2);
  _Float16* Qh    = (_Float16*)alloc((size_t)SEQ_L * D_MODEL * 2);
  _Float16* Kh    = (_Float16*)alloc((size_t)SEQ_L * D_MODEL * 2);
  _Float16* Vt    = (_Float16*)alloc((size_t)D_MODEL * SEQ_L * 2);  // [D,L]
  _Float16* AttnH = (_Float16*)alloc((size_t)SEQ_L * D_MODEL * 2);

  int nX = SEQ_L * D_MODEL;
  cvt_f32_to_f16<<<(nX + 255) / 256, 256, 0, stream>>>(X, Xh, nX);
  int nW = D_MODEL * D_MODEL;
  int wblk = (nW + 255) / 256;
  cvt_transpose<<<wblk, 256, 0, stream>>>(Wq, WqT, D_MODEL, D_MODEL);
  cvt_transpose<<<wblk, 256, 0, stream>>>(Wk, WkT, D_MODEL, D_MODEL);
  cvt_transpose<<<wblk, 256, 0, stream>>>(Wv, WvT, D_MODEL, D_MODEL);
  cvt_transpose<<<wblk, 256, 0, stream>>>(Wo, WoT, D_MODEL, D_MODEL);

  // 32x64 wave tiles: (2048/32)*(1024/64) = 1024 wave tiles, 8 waves/block
  int wtiles = (SEQ_L / 32) * (D_MODEL / 64);
  int gblocks = wtiles / 8;
  gemm16<<<gblocks, 256, 0, stream>>>(Xh, WqT, bq, Qh, nullptr, SEQ_L, D_MODEL, D_MODEL, 0);
  gemm16<<<gblocks, 256, 0, stream>>>(Xh, WkT, bk, Kh, nullptr, SEQ_L, D_MODEL, D_MODEL, 0);
  gemm16<<<gblocks, 256, 0, stream>>>(Xh, WvT, bv, Vt, nullptr, SEQ_L, D_MODEL, D_MODEL, 1);

  dim3 agrid(SEQ_L / 16, N_HEADS);
  arrow_attn<<<agrid, 32, 0, stream>>>(Qh, Kh, Vt, amask, AttnH);
  row0_fix<<<N_HEADS, 256, 0, stream>>>(Qh, Kh, Vt, amask, AttnH);

  gemm16<<<gblocks, 256, 0, stream>>>(AttnH, WoT, bo, nullptr, out, SEQ_L, D_MODEL, D_MODEL, 2);
}
